// LongConvolution_74380243632398
// MI455X (gfx1250) — matrix operand order repaired and checked by reading, not compiled, not run on hardware
//
#include <hip/hip_runtime.h>

typedef __attribute__((ext_vector_type(16))) __bf16 v16bf;
typedef __attribute__((ext_vector_type(8)))  float  v8f;

constexpr int SEQ = 8192;            // l
constexpr int DM  = 1024;            // d (channels)
constexpr int NB  = 4;               // batch
constexpr int PAD = 96;              // front zero pad (covers j >= -3 ramp: 3*32)
constexpr int PL  = PAD + SEQ + 32;  // 8320 padded line length in elements

static __device__ __forceinline__ unsigned short f2bf(float f) {
  unsigned int u = __builtin_bit_cast(unsigned int, f);
  u += 0x7fffu + ((u >> 16) & 1u);   // round-to-nearest-even fp32 -> bf16
  return (unsigned short)(u >> 16);
}

// ---------- phase 1a: x (b,l,d) f32 -> ub[c][b][s] bf16, front/back zero-padded ----------
__global__ void build_u(const float* __restrict__ x, unsigned short* __restrict__ ub) {
  __shared__ float tile[32][33];
  const int c0 = blockIdx.x * 32, s0 = blockIdx.y * 32, b = blockIdx.z;
  const int tx = threadIdx.x, ty = threadIdx.y;          // 32 x 8 block
#pragma unroll
  for (int yy = 0; yy < 4; ++yy) {
    int si = ty + 8 * yy;
    tile[si][tx] = x[((size_t)b * SEQ + (s0 + si)) * DM + (c0 + tx)];
  }
  __syncthreads();
#pragma unroll
  for (int yy = 0; yy < 4; ++yy) {
    int ci = ty + 8 * yy;
    ub[((size_t)(c0 + ci) * NB + b) * PL + PAD + (s0 + tx)] = f2bf(tile[tx][ci]);
  }
}

__global__ void zero_upad(unsigned short* __restrict__ ub) {
  int idx = blockIdx.x * blockDim.x + threadIdx.x;       // DM*NB*128 threads
  int cb = idx >> 7, r = idx & 127;
  int xx = (r < PAD) ? r : (SEQ + r);                    // [0,96) U [8288,8320)
  ub[(size_t)cb * PL + xx] = 0;
}

// ---------- phase 1b: reversed, zero-padded bf16 filter ----------
// kr[c][x] = kfull[(PAD+SEQ-1) - x],  kfull[i] = filt[c][i] for 0<=i<SEQ else 0
__global__ void build_krev(const float* __restrict__ filt, unsigned short* __restrict__ kr) {
  const int c = blockIdx.x;
  int xx = blockIdx.y * blockDim.x + threadIdx.x;
  if (xx >= PL) return;
  int i = (PAD + SEQ - 1) - xx;
  unsigned short v = 0;
  if (i >= 0 && i < SEQ) v = f2bf(filt[(size_t)c * SEQ + i]);
  kr[(size_t)c * PL + xx] = v;
}

// ---------- phase 2: Toeplitz x signal via v_wmma_f32_16x16x32_bf16 ----------
union ABFrag { uint4 q[2]; v16bf v; };

static __device__ __forceinline__ uint4 load16_ua(const unsigned short* p) {
  uint4 r;
  __builtin_memcpy(&r, p, 16);                           // hw-unaligned 16B load
  return r;
}

static __device__ __forceinline__ v8f wmma_step(const unsigned short* krc,
                                                const unsigned short* ubc,
                                                int abase, int bbase, int j,
                                                v8f acc) {
  // A fragment: A[i,s'] = k[dd+i-s'] from reversed filter; dd = dd0 - 32j folded into abase
  ABFrag a;
  const unsigned short* pa = krc + (abase + 32 * j);
  a.q[0] = load16_ua(pa);                                // elements 0..7 : K = e + 8*half
  a.q[1] = load16_ua(pa + 16);                           // elements 8..15: K = 16 + e' + 8*half
  // B fragment: column (4*mg+batch) = u[batch][32*(j+mg) + K], K = e + 8*half (e<8), 16 + e' + 8*half (e>=8)
  ABFrag bf;
  const unsigned short* pb = ubc + (bbase + 32 * j);     // 16B-aligned
  bf.q[0] = *(const uint4*)(pb);
  bf.q[1] = *(const uint4*)(pb + 16);
  return __builtin_amdgcn_wmma_f32_16x16x32_bf16(false, a.v, false, bf.v,
                                                 (short)0, acc, false, false);
}

__global__ void __launch_bounds__(256)
conv_wmma(const unsigned short* __restrict__ ub, const unsigned short* __restrict__ kr,
          float* __restrict__ out) {
  const int c    = blockIdx.x;
  const int lane = threadIdx.x & 31;
  // force wave id scalar: loop bounds become SALU-controlled and EXEC provably all-ones
  const int wave = __builtin_amdgcn_readfirstlane((int)(threadIdx.x >> 5));
  const int m    = lane & 15;                            // A row / B,D column
  const int half = lane >> 4;
  const int batch = m & 3;                               // column -> (s-group, batch)
  const int mg    = m >> 2;
  const unsigned short* krc = kr + (size_t)c * PL;
  const unsigned short* ubc = ub + ((size_t)c * NB + batch) * PL;
  const int albase = PAD + (SEQ - 1) - m + 8 * half;     // per-lane A base (minus dd)
  const int blbase = PAD + 32 * mg + 8 * half;           // per-lane B base (plus 32j)

  for (int q = 0; q < 4; ++q) {
    const int G  = blockIdx.y * 32 + wave + 8 * q;       // 128 groups per channel
    const int i0 = 8 * (G >> 1) + (G & 1);               // chunks {i0,i0+2,i0+4,i0+6}
    const int J0 = (16 * i0 + 15) >> 5;                  // scalar trip bound
    const int abase = albase - 16 * i0;                  // xa = abase + 32*j
    v8f acc0 = {0.f, 0.f, 0.f, 0.f, 0.f, 0.f, 0.f, 0.f};
    v8f acc1 = {0.f, 0.f, 0.f, 0.f, 0.f, 0.f, 0.f, 0.f};

    // dual accumulators: two independent WMMA chains (trip count J0+4 >= 4)
    int j = -3;
    for (; j + 1 <= J0; j += 2) {
      acc0 = wmma_step(krc, ubc, abase, blbase, j,     acc0);
      acc1 = wmma_step(krc, ubc, abase, blbase, j + 1, acc1);
    }
    if (j <= J0)
      acc0 = wmma_step(krc, ubc, abase, blbase, j, acc0);

    const v8f acc = acc0 + acc1;
    // D: row M = r + 8*half -> t = 16*(i0+2*mg) + 8*half + r, col -> (mg,batch)
    const int t0 = 16 * (i0 + 2 * mg) + 8 * half;
    float* o = out + ((size_t)batch * SEQ + t0) * DM + c;
#pragma unroll
    for (int r = 0; r < 8; ++r) o[(size_t)r * DM] = acc[r];
  }
}

// ---------- fallback (only if workspace is too small) ----------
__global__ void naive_conv(const float* __restrict__ x, const float* __restrict__ f,
                           float* __restrict__ out) {
  const int c = blockIdx.x * blockDim.x + threadIdx.x;
  const int t = blockIdx.y, b = blockIdx.z;
  float acc = 0.f;
  for (int s = 0; s <= t; ++s)
    acc += x[((size_t)b * SEQ + s) * DM + c] * f[(size_t)c * SEQ + (t - s)];
  out[((size_t)b * SEQ + t) * DM + c] = acc;
}

extern "C" void kernel_launch(void* const* d_in, const int* in_sizes, int n_in,
                              void* d_out, int out_size, void* d_ws, size_t ws_size,
                              hipStream_t stream) {
  const float* x    = (const float*)d_in[0];   // (4, 8192, 1024) f32
  const float* filt = (const float*)d_in[1];   // (1024, 8192)    f32
  float* out = (float*)d_out;                  // (4, 8192, 1024) f32

  const size_t u_elems = (size_t)DM * NB * PL;            // 34,078,720
  const size_t k_elems = (size_t)DM * PL;                 //  8,519,680
  const size_t need = (u_elems + k_elems) * sizeof(unsigned short);  // ~85.2 MB (L2-resident)

  if (ws_size >= need) {
    unsigned short* ubp = (unsigned short*)d_ws;
    unsigned short* krp = ubp + u_elems;
    build_u   <<<dim3(DM / 32, SEQ / 32, NB), dim3(32, 8), 0, stream>>>(x, ubp);
    zero_upad <<<dim3((DM * NB * 128) / 256), dim3(256),   0, stream>>>(ubp);
    build_krev<<<dim3(DM, (PL + 255) / 256), dim3(256),    0, stream>>>(filt, krp);
    conv_wmma <<<dim3(DM, 4), dim3(256), 0, stream>>>(ubp, krp, out);
  } else {
    naive_conv<<<dim3(DM / 256, SEQ, NB), dim3(256), 0, stream>>>(x, filt, out);
  }
}
